// MinCutPool_17824114278985
// MI455X (gfx1250) — matrix operand order repaired and checked by reading, not compiled, not run on hardware
//
#include <hip/hip_runtime.h>
#include <hip/hip_bf16.h>
#include <math.h>

// Problem constants (from reference)
#define BB   4
#define NN   4096
#define FF   128
#define KK   32
#define EPSF 1e-8f
#define MSPLIT 4   // split the m (reduction) dimension across waves for HBM concurrency

typedef __attribute__((ext_vector_type(16))) __bf16 v16bf;
typedef __attribute__((ext_vector_type(8)))  float  v8f;

// Native fp32 global atomic add (GLOBAL_ATOMIC_ADD_F32) instead of CAS loop.
static __device__ __forceinline__ void atomAddF(float* p, float v) {
#if defined(__AMDGCN__)
  unsafeAtomicAdd(p, v);
#else
  atomicAdd(p, v);
#endif
}

// ---------------------------------------------------------------------------
// Kernel 1: s_soft = softmax(logits/t2) per row; q[n] = sum_k s_soft^2;
// also write s_soft in bf16 pre-swizzled into the WMMA B-operand layout:
//   dst[ ((mt*32 + lane)*2 + half)*16 + (n&15) ]  with lane=(k&15)|(n&16), half=k>>4
// ---------------------------------------------------------------------------
__global__ void __launch_bounds__(256) k_soft(const float* __restrict__ logits,
                                              float* __restrict__ s_soft,
                                              float* __restrict__ q,
                                              __bf16* __restrict__ ssw,
                                              float inv_t2) {
  int n = blockIdx.x * blockDim.x + threadIdx.x;
  if (n >= NN) return;
  float v[KK];
  float mx = -1e30f;
#pragma unroll
  for (int k = 0; k < KK; ++k) { v[k] = logits[n * KK + k] * inv_t2; mx = fmaxf(mx, v[k]); }
  float s = 0.f;
#pragma unroll
  for (int k = 0; k < KK; ++k) { v[k] = __expf(v[k] - mx); s += v[k]; }
  float inv = 1.0f / s;
  float qq = 0.f;
  int mt = n >> 5, i = n & 15, hb = n & 16;
#pragma unroll
  for (int k = 0; k < KK; ++k) {
    float p = v[k] * inv;
    s_soft[n * KK + k] = p;
    qq += p * p;
    int hf = k >> 4;
    int lane = (k & 15) | hb;
    ssw[(((size_t)mt * 32 + lane) * 2 + hf) * 16 + i] = (__bf16)p;
  }
  q[n] = qq;
}

// ---------------------------------------------------------------------------
// Kernel 2: argmax over logits/t1 + gumbel (== argmax of y_soft);
// write s_sample = one_hot + EPS (forward value of straight-through),
// plus am[] and the bf16 swizzled s_sample for the WMMA B-operand.
// ---------------------------------------------------------------------------
__global__ void __launch_bounds__(256) k_sample(const float* __restrict__ logits,
                                                const float* __restrict__ gum,
                                                float* __restrict__ s_sample,
                                                int* __restrict__ am,
                                                __bf16* __restrict__ ssmw,
                                                float inv_t1) {
  int id = blockIdx.x * blockDim.x + threadIdx.x;   // id = b*NN + n
  if (id >= BB * NN) return;
  int b = id / NN, n = id - b * NN;
  int best = 0; float bv = -1e30f;
#pragma unroll
  for (int k = 0; k < KK; ++k) {
    float z = logits[n * KK + k] * inv_t1 + gum[(size_t)id * KK + k];
    if (z > bv) { bv = z; best = k; }
  }
  am[id] = best;
  int mt = n >> 5, i = n & 15, hb = n & 16;
  size_t base = (size_t)b * NN * KK;
#pragma unroll
  for (int k = 0; k < KK; ++k) {
    float val = (k == best) ? (1.0f + EPSF) : EPSF;
    s_sample[(size_t)id * KK + k] = val;
    int hf = k >> 4;
    int lane = (k & 15) | hb;
    ssmw[base + (((size_t)mt * 32 + lane) * 2 + hf) * 16 + i] = (__bf16)val;
  }
}

// ---------------------------------------------------------------------------
// Kernel 3 (main): one wave per (16-row strip, m-chunk) of adj[b]. Fused pass:
//   As  = adj @ s_soft   (bf16 WMMA, f32 acc)     -> reduced into mincut_num
//   As2 = adj @ s_sample (bf16 WMMA, f32 acc)     -> scattered into out_adj buckets
//   d   = adj row-sums                            -> reduced into mincut_den
// adj is read exactly once from HBM; MSPLIT m-chunks give 4096 concurrent
// waves so enough b128 loads are in flight to saturate the 23.3 TB/s pipe.
// All epilogue consumers are additive, so chunk-partial accumulators combine
// correctly through atomics.
// ---------------------------------------------------------------------------
__global__ void __launch_bounds__(128) k_gemm(const float* __restrict__ adj,
                                              const float* __restrict__ s_soft,
                                              const float* __restrict__ q,
                                              const int* __restrict__ am,
                                              const __bf16* __restrict__ ssw,
                                              const __bf16* __restrict__ ssmw,
                                              float* __restrict__ num_acc,
                                              float* __restrict__ den_acc,
                                              float* __restrict__ bucket,
                                              float* __restrict__ colsum) {
  const int wave = (blockIdx.x * blockDim.x + threadIdx.x) >> 5;
  const int lane = threadIdx.x & 31;
  const int nwaves = BB * (NN / 16) * MSPLIT;
  if (wave >= nwaves) return;                       // wave-uniform, EXEC stays full
  const int rbid  = wave / MSPLIT;                  // (b, rowblock)
  const int chunk = wave - rbid * MSPLIT;           // m-chunk
  const int b  = rbid / (NN / 16);
  const int rb = rbid - b * (NN / 16);
  const int rowbase = rb * 16;
  const int m0 = chunk * (NN / MSPLIT);
  const int m1 = m0 + (NN / MSPLIT);

  // A operand addressing (contiguous k order, same as the B packing):
  //   lane<16 : row=rowbase+lane,    K = m+{0..15}
  //   lane>=16: row=rowbase+lane-16, K = m+{16..31}
  const int arow = rowbase + (lane & 15);
  const float* __restrict__ arowp = adj + (size_t)b * NN * NN + (size_t)arow * NN;
  const int cb = (lane & 16) ? 16 : 0;

  const __bf16* __restrict__ bs = ssw;                                   // [mt][lane][2][16]
  const __bf16* __restrict__ bm = ssmw + (size_t)b * NN * KK;            // per-batch

  v8f acc_soft_lo = {}, acc_soft_hi = {}, acc_samp_lo = {}, acc_samp_hi = {};
  float rsum = 0.f;

  for (int m = m0; m < m1; m += 32) {
    const float4* p0 = (const float4*)(arowp + m + cb);
    const float4* p1 = (const float4*)(arowp + m + cb + 8);
    float4 f0 = p0[0];
    float4 f1 = p0[1];
    float4 f2 = p1[0];
    float4 f3 = p1[1];
    if (m + 32 < m1) __builtin_prefetch(arowp + m + 32 + cb, 0, 1);  // global_prefetch_b8

    v16bf a;
    a[0]  = (__bf16)f0.x; a[1]  = (__bf16)f0.y; a[2]  = (__bf16)f0.z; a[3]  = (__bf16)f0.w;
    a[4]  = (__bf16)f1.x; a[5]  = (__bf16)f1.y; a[6]  = (__bf16)f1.z; a[7]  = (__bf16)f1.w;
    a[8]  = (__bf16)f2.x; a[9]  = (__bf16)f2.y; a[10] = (__bf16)f2.z; a[11] = (__bf16)f2.w;
    a[12] = (__bf16)f3.x; a[13] = (__bf16)f3.y; a[14] = (__bf16)f3.z; a[15] = (__bf16)f3.w;

    rsum += (f0.x + f0.y + f0.z + f0.w) + (f1.x + f1.y + f1.z + f1.w)
          + (f2.x + f2.y + f2.z + f2.w) + (f3.x + f3.y + f3.z + f3.w);

    size_t boff = ((size_t)(m >> 5) * 32 + lane) * 32;   // *2 halves *16 elems
    v16bf b_soft_lo = *(const v16bf*)(bs + boff);
    v16bf b_soft_hi = *(const v16bf*)(bs + boff + 16);
    v16bf b_samp_lo = *(const v16bf*)(bm + boff);
    v16bf b_samp_hi = *(const v16bf*)(bm + boff + 16);

    acc_soft_lo = __builtin_amdgcn_wmma_f32_16x16x32_bf16(false, a, false, b_soft_lo,
                                                          (short)0, acc_soft_lo, false, false);
    acc_soft_hi = __builtin_amdgcn_wmma_f32_16x16x32_bf16(false, a, false, b_soft_hi,
                                                          (short)0, acc_soft_hi, false, false);
    acc_samp_lo = __builtin_amdgcn_wmma_f32_16x16x32_bf16(false, a, false, b_samp_lo,
                                                          (short)0, acc_samp_lo, false, false);
    acc_samp_hi = __builtin_amdgcn_wmma_f32_16x16x32_bf16(false, a, false, b_samp_hi,
                                                          (short)0, acc_samp_hi, false, false);
  }

  // Partial row sums: lane L and lane L^16 together cover this chunk's columns.
  float drow = rsum + __shfl_xor(rsum, 16, 32);
  float den_part = (lane < 16) ? drow * q[rowbase + lane] : 0.f;

  // C/D layout: VGPR r, lane<16 -> M=r ; lane>=16 -> M=r+8 ; N = lane&15.
  const int rowoff = rowbase + ((lane & 16) >> 1);
  const int col = lane & 15;

  float num_part = 0.f;
#pragma unroll
  for (int r = 0; r < 8; ++r) {
    const float* sp = s_soft + (size_t)(rowoff + r) * KK;
    num_part += acc_soft_lo[r] * sp[col] + acc_soft_hi[r] * sp[col + 16];
  }
#pragma unroll
  for (int off = 16; off > 0; off >>= 1) {
    num_part += __shfl_xor(num_part, off, 32);
    den_part += __shfl_xor(den_part, off, 32);
  }
  if (lane == 0) {
    atomAddF(&num_acc[b], num_part);
    atomAddF(&den_acc[b], den_part);
  }

  // EPS term of out_adj: wave-reduce per-column sums first (2 atomics/lane-half).
  float clo = 0.f, chi = 0.f;
#pragma unroll
  for (int r = 0; r < 8; ++r) { clo += acc_samp_lo[r]; chi += acc_samp_hi[r]; }
  clo += __shfl_xor(clo, 16, 32);
  chi += __shfl_xor(chi, 16, 32);
  if (lane < 16) {
    atomAddF(&colsum[b * KK + lane], clo);
    atomAddF(&colsum[b * KK + 16 + lane], chi);
  }

  // One-hot part of out_adj = s_sample^T As2 : scatter rows into bucket[am[row]].
#pragma unroll
  for (int r = 0; r < 8; ++r) {
    int row = rowoff + r;
    int k = am[b * NN + row];
    atomAddF(&bucket[((size_t)b * KK + k) * KK + col], acc_samp_lo[r]);
    atomAddF(&bucket[((size_t)b * KK + k) * KK + col + 16], acc_samp_hi[r]);
  }
}

// ---------------------------------------------------------------------------
// Kernel 4: out_emb[b,k,f] = sum_{n: am=k} emb + EPS * sum_n emb   (LDS atomics)
// ---------------------------------------------------------------------------
__global__ void __launch_bounds__(512) k_emb(const float* __restrict__ emb,
                                             const int* __restrict__ am,
                                             float* __restrict__ out_emb) {
  const int b = blockIdx.x;
  __shared__ float acc[KK * FF];          // 16 KB of the 320KB WGP LDS
  int tid = threadIdx.x;
  for (int i = tid; i < KK * FF; i += blockDim.x) acc[i] = 0.f;
  __syncthreads();
  int f = tid & (FF - 1);
  int slice = tid >> 7;                   // 4 n-slices
  for (int n = slice * (NN / 4); n < (slice + 1) * (NN / 4); ++n) {
    float e = emb[((size_t)b * NN + n) * FF + f];
    atomicAdd(&acc[am[b * NN + n] * FF + f], e);
  }
  __syncthreads();
  for (int i = tid; i < KK * FF; i += blockDim.x) {
    int ff = i & (FF - 1);
    float cs = 0.f;
#pragma unroll
    for (int k = 0; k < KK; ++k) cs += acc[k * FF + ff];   // sum_k acc == sum_n emb
    out_emb[(size_t)b * KK * FF + i] = acc[i] + EPSF * cs;
  }
}

// ---------------------------------------------------------------------------
// Kernel 5: finalize out_adj (zero diag + symmetric degree normalization)
// and compute mincut_loss. One wave per output row (blockDim=1024 -> 32 waves).
// ---------------------------------------------------------------------------
__global__ void __launch_bounds__(1024) k_adj(const float* __restrict__ bucket,
                                              const float* __restrict__ colsum,
                                              const float* __restrict__ num_acc,
                                              const float* __restrict__ den_acc,
                                              float* __restrict__ out_adj,
                                              float* __restrict__ loss_mc) {
  const int b = blockIdx.x;
  const int k = threadIdx.x >> 5;
  const int l = threadIdx.x & 31;
  float v = bucket[((size_t)b * KK + k) * KK + l] + EPSF * colsum[b * KK + l];
  if (k == l) v = 0.f;
  float rs = v;
#pragma unroll
  for (int off = 16; off > 0; off >>= 1) rs += __shfl_xor(rs, off, 32);
  float dk = sqrtf(rs) + EPSF;
  __shared__ float dsh[KK];
  if (l == 0) dsh[k] = dk;
  __syncthreads();
  out_adj[((size_t)b * KK + k) * KK + l] = v / dk / dsh[l];
  if (b == 0 && threadIdx.x == 0) {
    float s = 0.f;
#pragma unroll
    for (int bb = 0; bb < BB; ++bb) s += num_acc[bb] / den_acc[bb];
    loss_mc[0] = -s / (float)BB;
  }
}

// ---------------------------------------------------------------------------
// Kernel 6: ortho loss on the 32x32 gram matrix of s_soft.
// ---------------------------------------------------------------------------
__global__ void __launch_bounds__(1024) k_ortho(const float* __restrict__ s_soft,
                                                float* __restrict__ loss_or) {
  const int i = threadIdx.x >> 5;
  const int j = threadIdx.x & 31;
  float ss = 0.f;
  for (int n = 0; n < NN; ++n)
    ss += s_soft[n * KK + i] * s_soft[n * KK + j];
  __shared__ float red[32];
  __shared__ float normsh;
  float sq = ss * ss;
#pragma unroll
  for (int off = 16; off > 0; off >>= 1) sq += __shfl_xor(sq, off, 32);
  if (j == 0) red[i] = sq;
  __syncthreads();
  if (threadIdx.x == 0) {
    float t = 0.f;
#pragma unroll
    for (int w = 0; w < 32; ++w) t += red[w];
    normsh = sqrtf(t);
  }
  __syncthreads();
  float inorm = 1.0f / normsh;
  float diag = (i == j) ? rsqrtf((float)KK) : 0.f;
  float d = ss * inorm - diag;
  d = d * d;
#pragma unroll
  for (int off = 16; off > 0; off >>= 1) d += __shfl_xor(d, off, 32);
  if (j == 0) red[i] = d;
  __syncthreads();
  if (threadIdx.x == 0) {
    float t = 0.f;
#pragma unroll
    for (int w = 0; w < 32; ++w) t += red[w];
    loss_or[0] = sqrtf(t);
  }
}

// ---------------------------------------------------------------------------
extern "C" void kernel_launch(void* const* d_in, const int* in_sizes, int n_in,
                              void* d_out, int out_size, void* d_ws, size_t ws_size,
                              hipStream_t stream) {
  const float* emb    = (const float*)d_in[0];
  const float* adj    = (const float*)d_in[1];
  const float* logits = (const float*)d_in[2];
  const float* gum    = (const float*)d_in[3];

  // Outputs, flat concat in return order
  float* out_emb  = (float*)d_out;                         // B*K*F
  float* out_adj  = out_emb + BB * KK * FF;                // B*K*K
  float* s_sample = out_adj + BB * KK * KK;                // B*N*K
  float* loss_mc  = s_sample + (size_t)BB * NN * KK;       // 1
  float* loss_or  = loss_mc + 1;                           // 1

  // Workspace layout (accumulators first so one memset clears them)
  float*  num_acc = (float*)d_ws;                          // B
  float*  den_acc = num_acc + BB;                          // B
  float*  bucket  = den_acc + BB;                          // B*K*K
  float*  colsum  = bucket + BB * KK * KK;                 // B*K
  size_t  zfloats = (size_t)(BB + BB + BB * KK * KK + BB * KK);
  float*  s_soft  = colsum + BB * KK;                      // N*K
  float*  q       = s_soft + (size_t)NN * KK;              // N
  int*    am      = (int*)(q + NN);                        // B*N
  __bf16* ssw     = (__bf16*)(am + BB * NN);               // N*K  (swizzled bf16)
  __bf16* ssmw    = ssw + (size_t)NN * KK;                 // B*N*K (swizzled bf16)

  hipMemsetAsync(d_ws, 0, zfloats * sizeof(float), stream);

  const float t1 = 0.99995f;                 // max(1.0*0.99995, 0.05)
  const float t2 = 0.99995f * 0.99995f;      // max(t1*0.99995, 0.05)

  k_soft  <<<(NN + 255) / 256, 256, 0, stream>>>(logits, s_soft, q, ssw, 1.0f / t2);
  k_sample<<<(BB * NN + 255) / 256, 256, 0, stream>>>(logits, gum, s_sample, am, ssmw, 1.0f / t1);

  // B*(N/16)*MSPLIT = 4096 waves, 4 waves (128 threads) per block -> 1024 blocks
  k_gemm<<<(BB * (NN / 16) * MSPLIT) / 4, 128, 0, stream>>>(adj, s_soft, q, am, ssw, ssmw,
                                                            num_acc, den_acc, bucket, colsum);

  k_emb  <<<BB, 512, 0, stream>>>(emb, am, out_emb);
  k_adj  <<<BB, 1024, 0, stream>>>(bucket, colsum, num_acc, den_acc, out_adj, loss_mc);
  k_ortho<<<1, 1024, 0, stream>>>(s_soft, loss_or);
}
